// MHCA_57363583206030
// MI455X (gfx1250) — matrix operand order repaired and checked by reading, not compiled, not run on hardware
//
#include <hip/hip_runtime.h>
#include <hip/hip_bf16.h>

// ---------------------------------------------------------------------------
// MHCA on MI455X (gfx1250, wave32, WMMA).
// Bottleneck: writing attn [4,4096,4096] f32 = 268MB -> ~11.5us @ 23.3TB/s.
// Per 16-row block keep the whole 16x4096 energy slab in LDS (16*4100*4 =
// 256KB, CDNA5 allows 320KB/WG), fuse QK^T (wmma f16), softmax, one coalesced
// non-temporal b128 attn store, and PV (wmma f16) in one kernel.
// All WMMA loops have compile-time trip counts (EXEC stays all-ones, no
// exec-mask loop control); two independent accumulation chains per iteration
// hide the WMMA->WMMA RAW hazard slots.
// ---------------------------------------------------------------------------

typedef __attribute__((ext_vector_type(16))) _Float16 v16h;
typedef __attribute__((ext_vector_type(8)))  float    v8f;
typedef __attribute__((ext_vector_type(4)))  float    v4f;

#define BATCH 4
#define N1    4096      // H1*W1
#define C1    64        // D1
#define D2C   32        // D2
#define ESTR  4100      // padded LDS row stride (floats): 16B-aligned rows,
                        // bank = (4*row + col) % 64 -> conflict-free columns

// ---------------- positional encodings (elementwise, analytic) -------------

__global__ void pex_kernel(const float* __restrict__ x, float* __restrict__ xout) {
    int idx = blockIdx.x * blockDim.x + threadIdx.x;
    if (idx >= BATCH * 64 * 64 * 64) return;
    int w = idx & 63, h = (idx >> 6) & 63, c = (idx >> 12) & 63;
    const float F = -9.210340371976184f / 32.0f;     // -ln(1e4)/dm, dm=32
    int j = (c & 31) >> 1;
    float pos = (c < 32) ? (float)w : (float)h;
    float arg = pos * __expf((float)(2 * j) * F);
    float pe = (c & 1) ? __cosf(arg) : __sinf(arg);
    xout[idx] = x[idx] + pe;
}

__global__ void peb_kernel(const float* __restrict__ bi, float* __restrict__ bpe) {
    int idx = blockIdx.x * blockDim.x + threadIdx.x;
    if (idx >= BATCH * 32 * 128 * 128) return;
    int w = idx & 127, h = (idx >> 7) & 127, c = (idx >> 14) & 31;
    const float F = -9.210340371976184f / 16.0f;     // dm=16
    int j = (c & 15) >> 1;
    float pos = (c < 16) ? (float)w : (float)h;
    float arg = pos * __expf((float)(2 * j) * F);
    float pe = (c & 1) ? __cosf(arg) : __sinf(arg);
    bpe[idx] = bi[idx] + pe;
}

// ---------------- Q,K projection (LDS-tiled, outputs f16) ------------------
// q[b,n,o] = sum_c Wq[o,c]*xpe[b,c,n] ; kt[b,m,o] = sum_c Wk[o,c]*xpe[b,c,m]

__global__ __launch_bounds__(64) void qk_kernel(
        const float* __restrict__ xpe, const float* __restrict__ Wq,
        const float* __restrict__ Wk, _Float16* __restrict__ qf,
        _Float16* __restrict__ kf) {
    __shared__ float lwq[64 * 65];   // padded: bank = (o+c)%64, conflict-free
    __shared__ float lwk[64 * 65];
    __shared__ float lx[64 * 64];
    int t = threadIdx.x;             // t == output channel o
    int b = blockIdx.y;
    int n0 = blockIdx.x * 64;
    for (int i = 0; i < 64; ++i) {
        int idx = i * 64 + t;
        lwq[(idx >> 6) * 65 + (idx & 63)] = Wq[idx];
        lwk[(idx >> 6) * 65 + (idx & 63)] = Wk[idx];
    }
    for (int c = 0; c < 64; ++c)
        lx[c * 64 + t] = xpe[((size_t)b * 64 + c) * N1 + n0 + t];  // coalesced
    __syncthreads();
    for (int nn = 0; nn < 64; ++nn) {
        float aq = 0.f, ak = 0.f;
        for (int c = 0; c < 64; ++c) {
            float xv = lx[c * 64 + nn];             // broadcast
            aq += lwq[t * 65 + c] * xv;
            ak += lwk[t * 65 + c] * xv;
        }
        size_t o = ((size_t)b * N1 + n0 + nn) * 64 + t;
        qf[o] = (_Float16)aq;
        kf[o] = (_Float16)ak;
    }
}

// ---------------- V projection (strided 1x1 conv), stored TRANSPOSED -------
// vT[b,d,m] = sum_c Wv[d,c]*bpe[b,c,2h,2w]  (m = h*64+w) -> contiguous-K
// B-fragments for the PV WMMA become single 32B loads.

__global__ __launch_bounds__(256) void v_kernel(
        const float* __restrict__ bpe, const float* __restrict__ Wv,
        _Float16* __restrict__ vT) {
    __shared__ float lwv[32 * 33];
    int tid = threadIdx.x;
    int b = blockIdx.y;
    int n0 = blockIdx.x * 8;
    for (int i = tid; i < 1024; i += 256)
        lwv[(i >> 5) * 33 + (i & 31)] = Wv[i];
    __syncthreads();
    int d = tid >> 3, nn = tid & 7;
    int n = n0 + nn, h = n >> 6, w = n & 63;
    const float* bp = bpe + (size_t)b * 32 * 16384 + (2 * h) * 128 + 2 * w;
    float acc = 0.f;
    for (int c = 0; c < 32; ++c)
        acc += lwv[d * 33 + c] * bp[(size_t)c * 16384];
    vT[((size_t)b * 32 + d) * N1 + n] = (_Float16)acc;
}

// ---------------- fused attention: QK^T + softmax + attn store + PV --------

#define ATTN_SMEM ((16 * ESTR + 16 + 16 + 256 + 8 * 512) * 4)  // 279,936 B

__global__ __launch_bounds__(256) void attn_kernel(
        const _Float16* __restrict__ qf, const _Float16* __restrict__ kf,
        const _Float16* __restrict__ vT, float* __restrict__ attn_out,
        float* __restrict__ out_attn) {
    extern __shared__ char smem_raw[];
    float* eS     = (float*)smem_raw;          // [16][4100] energy/prob slab
    float* rowmax = eS + 16 * ESTR;            // [16]
    float* rowinv = rowmax + 16;               // [16]
    float* red    = rowinv + 16;               // [16][16] partials
    float* cpart  = red + 256;                 // [8 waves][2 tiles][256]

    const int tid  = threadIdx.x;
    const int lane = tid & 31;
    const int wave = tid >> 5;                 // 0..7
    const int b    = blockIdx.y;
    const int nb   = blockIdx.x * 16;          // row-block base
    const int mrow = lane & 15;
    const int hi   = lane >> 4;                // half-wave select

    // ---- Phase A: E(16 x 4096) = Q(16x64) * K^T(64x4096), wmma f16 k=32 ----
    // A-frag (contiguous K, matching B): lanes K in {hb..hb+15},
    // hb = hi*16.  Two frags cover C1=64.  Two independent m-tiles per
    // iteration -> two accumulation chains hide the WMMA RAW hazard.
    const _Float16* qrow = qf + ((size_t)b * N1 + nb + mrow) * 64;
    v16h a0, a1;
    {
        const int hb = hi * 16;
#pragma unroll
        for (int j = 0; j < 8; ++j) {
            a0[j]     = qrow[hb + j];
            a0[8 + j] = qrow[hb + 8 + j];
            a1[j]     = qrow[32 + hb + j];
            a1[8 + j] = qrow[32 + hb + 8 + j];
        }
    }
    const int kb = hi * 16;                    // B-frag: contiguous K chunk
    for (int it = 0; it < 16; ++it) {          // uniform trip count
        const int mtA = it * 16 + wave;
        const int mtB = mtA + 8;
        const _Float16* ktA = kf + ((size_t)b * N1 + mtA * 16 + mrow) * 64;
        const _Float16* ktB = kf + ((size_t)b * N1 + mtB * 16 + mrow) * 64;
        v16h bA0 = *(const v16h*)(ktA + kb);
        v16h bA1 = *(const v16h*)(ktA + 32 + kb);
        v16h bB0 = *(const v16h*)(ktB + kb);
        v16h bB1 = *(const v16h*)(ktB + 32 + kb);
        v8f cA = {}, cB = {};
        cA = __builtin_amdgcn_wmma_f32_16x16x32_f16(false, a0, false, bA0,
                                                    (short)0, cA, false, false);
        cB = __builtin_amdgcn_wmma_f32_16x16x32_f16(false, a0, false, bB0,
                                                    (short)0, cB, false, false);
        cA = __builtin_amdgcn_wmma_f32_16x16x32_f16(false, a1, false, bA1,
                                                    (short)0, cA, false, false);
        cB = __builtin_amdgcn_wmma_f32_16x16x32_f16(false, a1, false, bB1,
                                                    (short)0, cB, false, false);
        const int colA = mtA * 16 + mrow;      // C layout: N = lane&15
        const int colB = mtB * 16 + mrow;
#pragma unroll
        for (int r = 0; r < 8; ++r) {          // M = r + 8*hi
            eS[(r + hi * 8) * ESTR + colA] = cA[r];
            eS[(r + hi * 8) * ESTR + colB] = cB[r];
        }
    }
    __syncthreads();

    // ---- Phase B: rowwise softmax stats, slab -> exp(e - max) in place ----
    {
        const int row = tid & 15;
        const int wk  = tid >> 4;              // 16 workers per row
        float m = -3.4e38f;
        for (int i = 0; i < 256; ++i)          // uniform trip count
            m = fmaxf(m, eS[row * ESTR + wk + i * 16]);
        red[row * 16 + wk] = m;
        __syncthreads();
        if (wk == 0) {
            float mm = red[row * 16];
            for (int i = 1; i < 16; ++i) mm = fmaxf(mm, red[row * 16 + i]);
            rowmax[row] = mm;
        }
        __syncthreads();
        const float rmax = rowmax[row];
        float s = 0.f;
        for (int i = 0; i < 256; ++i) {
            const int c = wk + i * 16;
            float p = __expf(eS[row * ESTR + c] - rmax);
            eS[row * ESTR + c] = p;
            s += p;
        }
        red[row * 16 + wk] = s;
        __syncthreads();
        if (wk == 0) {
            float ss = 0.f;
            for (int i = 0; i < 16; ++i) ss += red[row * 16 + i];
            rowinv[row] = 1.0f / ss;
        }
        __syncthreads();
    }

    // ---- Phase C: coalesced NON-TEMPORAL b128 store of attn (256KB/block) -
    // attn is written once and never re-read: TH=NT keeps it out of L2 so
    // tconv/oa/bpe stay resident.
    {
        v4f* dst = (v4f*)(attn_out + ((size_t)b * N1 + nb) * N1);
        for (int i = 0; i < 64; ++i) {
            int e4  = i * 256 + tid;           // float4 index over [16][1024]
            int row = e4 >> 10;
            int col = (e4 & 1023) << 2;
            float ri = rowinv[row];
            v4f p = *(const v4f*)(&eS[row * ESTR + col]);
            p *= ri;
            __builtin_nontemporal_store(p, dst + e4);
        }
    }

    // ---- Phase D: out(16x32) = P(16x4096) * V(4096x32), wmma f16 ----------
    v8f acc0 = {}, acc1 = {};
    const float rinv = rowinv[mrow];
    for (int it = 0; it < 8; ++it) {           // uniform trip count
        const int mA = (it * 16 + wave) * 32;  // two 32-wide K chunks
        const int mB = mA + 8 * 32;
        const int hb = hi * 16;
        v16h pA, pB;                           // P in A-frag layout, from LDS
#pragma unroll
        for (int j = 0; j < 8; ++j) {
            pA[j]     = (_Float16)(eS[mrow * ESTR + mA + hb + j] * rinv);
            pA[8 + j] = (_Float16)(eS[mrow * ESTR + mA + hb + 8 + j] * rinv);
            pB[j]     = (_Float16)(eS[mrow * ESTR + mB + hb + j] * rinv);
            pB[8 + j] = (_Float16)(eS[mrow * ESTR + mB + hb + 8 + j] * rinv);
        }
        const _Float16* v0 = vT + ((size_t)b * 32 + mrow) * N1 + kb;
        const _Float16* v1 = vT + ((size_t)b * 32 + 16 + mrow) * N1 + kb;
        v16h bA0 = *(const v16h*)(v0 + mA);    // contiguous 32B (transposed V)
        v16h bA1 = *(const v16h*)(v1 + mA);
        v16h bB0 = *(const v16h*)(v0 + mB);
        v16h bB1 = *(const v16h*)(v1 + mB);
        acc0 = __builtin_amdgcn_wmma_f32_16x16x32_f16(false, pA, false, bA0,
                                                      (short)0, acc0, false, false);
        acc1 = __builtin_amdgcn_wmma_f32_16x16x32_f16(false, pA, false, bA1,
                                                      (short)0, acc1, false, false);
        acc0 = __builtin_amdgcn_wmma_f32_16x16x32_f16(false, pB, false, bB0,
                                                      (short)0, acc0, false, false);
        acc1 = __builtin_amdgcn_wmma_f32_16x16x32_f16(false, pB, false, bB1,
                                                      (short)0, acc1, false, false);
    }
#pragma unroll
    for (int r = 0; r < 8; ++r) {
        cpart[wave * 512 + r * 32 + lane]       = acc0[r];
        cpart[wave * 512 + 256 + r * 32 + lane] = acc1[r];
    }
    __syncthreads();
    for (int t2 = 0; t2 < 2; ++t2) {           // cross-wave C reduction
        int f = t2 * 256 + tid;
        float s = 0.f;
#pragma unroll
        for (int w = 0; w < 8; ++w) s += cpart[w * 512 + f];
        int tile = f >> 8;
        int r    = (f >> 5) & 7;
        int ln   = f & 31;
        int M    = r + ((ln >> 4) << 3);
        int Nc   = ln & 15;
        out_attn[((size_t)b * N1 + nb + M) * 32 + tile * 16 + Nc] = s;
    }
}

// ---------------- 1x1 conv (Wc,bc) + deterministic BN partials -------------
// z[b,c,hw] = out_attn raw-reinterpret: flat[b, c*4096+hw]

__global__ __launch_bounds__(256) void conv_kernel(
        const float* __restrict__ oa, const float* __restrict__ Wc,
        const float* __restrict__ bc, float* __restrict__ tconv,
        float* __restrict__ psum, float* __restrict__ psumsq) {
    __shared__ float lwc[32 * 33];
    __shared__ float rs[256];
    __shared__ float rs2[256];
    int tid = threadIdx.x;
    int b = blockIdx.y;
    int hw0 = blockIdx.x * 8;
    for (int i = tid; i < 1024; i += 256)
        lwc[(i >> 5) * 33 + (i & 31)] = Wc[i];
    __syncthreads();
    int o = tid & 31, hh = tid >> 5;
    int hw = hw0 + hh;
    const float* z = oa + (size_t)b * (32 * N1) + hw;   // z[c] stride 4096
    float acc = bc[o];
    for (int c = 0; c < 32; ++c)
        acc += lwc[o * 33 + c] * z[(size_t)c * N1];      // wave-broadcast read
    tconv[((size_t)b * 32 + o) * N1 + hw] = acc;
    rs[tid] = acc;
    rs2[tid] = acc * acc;
    __syncthreads();
    for (int s = 128; s >= 32; s >>= 1) {                // reduce over hh
        if (tid < s) { rs[tid] += rs[tid + s]; rs2[tid] += rs2[tid + s]; }
        __syncthreads();
    }
    if (tid < 32) {
        int blk = blockIdx.y * gridDim.x + blockIdx.x;
        psum[(size_t)blk * 32 + tid]   = rs[tid];
        psumsq[(size_t)blk * 32 + tid] = rs2[tid];
    }
}

__global__ __launch_bounds__(256) void bnfin_kernel(
        const float* __restrict__ psum, const float* __restrict__ psumsq,
        const float* __restrict__ gamma, const float* __restrict__ beta,
        float* __restrict__ bnAB) {
    __shared__ float rs[256];
    __shared__ float rs2[256];
    int tid = threadIdx.x;
    int o = tid & 31, ch = tid >> 5;
    float s = 0.f, s2 = 0.f;
    for (int i = ch; i < 2048; i += 8) {
        s += psum[(size_t)i * 32 + o];
        s2 += psumsq[(size_t)i * 32 + o];
    }
    rs[tid] = s;
    rs2[tid] = s2;
    __syncthreads();
    for (int st = 128; st >= 32; st >>= 1) {
        if (tid < st) { rs[tid] += rs[tid + st]; rs2[tid] += rs2[tid + st]; }
        __syncthreads();
    }
    if (tid < 32) {
        const float cnt = 16384.f;                       // B*H1*W1
        float mean = rs[tid] / cnt;
        float var  = rs2[tid] / cnt - mean * mean;       // biased, as jnp.var
        float A = gamma[tid] * rsqrtf(var + 1e-5f);
        bnAB[tid]      = A;
        bnAB[32 + tid] = beta[tid] - mean * A;
    }
}

// ---------------- BN-affine + sigmoid + bilinear(AC) x2 + multiply ---------

__global__ void final_kernel(const float* __restrict__ tconv,
                             const float* __restrict__ bnAB,
                             const float* __restrict__ bpe,
                             float* __restrict__ outp) {
    int idx = blockIdx.x * blockDim.x + threadIdx.x;
    if (idx >= BATCH * 32 * 128 * 128) return;
    int j = idx & 127, i = (idx >> 7) & 127;
    int o = (idx >> 14) & 31, b = idx >> 19;
    float A = bnAB[o], Bs = bnAB[32 + o];
    const float* tc = tconv + ((size_t)b * 32 + o) * N1;
    const float r = 63.0f / 127.0f;                      // align_corners=True
    float chf = i * r, cwf = j * r;
    int h0 = (int)chf, w0 = (int)cwf;
    float fh = chf - h0, fw = cwf - w0;
    int h1 = h0 + 1; if (h1 > 63) h1 = 63;
    int w1 = w0 + 1; if (w1 > 63) w1 = 63;
    auto sg = [&](int hh, int ww) {
        float v = tc[hh * 64 + ww] * A + Bs;
        return 1.0f / (1.0f + __expf(-v));
    };
    float top = sg(h0, w0) * (1.f - fw) + sg(h0, w1) * fw;
    float bot = sg(h1, w0) * (1.f - fw) + sg(h1, w1) * fw;
    float val = (top * (1.f - fh) + bot * fh) * bpe[idx];
    __builtin_nontemporal_store(val, &outp[idx]);
}

// ---------------------------------------------------------------------------

extern "C" void kernel_launch(void* const* d_in, const int* in_sizes, int n_in,
                              void* d_out, int out_size, void* d_ws, size_t ws_size,
                              hipStream_t stream) {
    (void)in_sizes; (void)n_in; (void)out_size; (void)ws_size;
    const float* x     = (const float*)d_in[0];
    const float* bin   = (const float*)d_in[1];
    const float* Wq    = (const float*)d_in[2];
    const float* Wk    = (const float*)d_in[3];
    const float* Wv    = (const float*)d_in[4];
    const float* Wc    = (const float*)d_in[5];
    const float* bc    = (const float*)d_in[6];
    const float* gamma = (const float*)d_in[7];
    const float* beta  = (const float*)d_in[8];

    float* out = (float*)d_out;
    // tuple layout: out[4,32,128,128] | attn[4,4096,4096] | x[4,64,64,64]
    float* out_final = out;
    float* attn_out  = out + 2097152;
    float* x_out     = out + 2097152 + 67108864;

    char* ws = (char*)d_ws;                      // ~18.4 MB total
    _Float16* qf   = (_Float16*)(ws + 0);        // [4,4096,64]  f16, 2 MB
    _Float16* kf   = (_Float16*)(ws + 2097152);  // [4,4096,64]  f16, 2 MB
    _Float16* vT   = (_Float16*)(ws + 4194304);  // [4,32,4096]  f16, 1 MB
    float* bpe     = (float*)(ws + 5242880);     // [4,32,128,128] 8 MB
    float* oa      = (float*)(ws + 13631488);    // [4,4096,32]  2 MB
    float* tconv   = (float*)(ws + 15728640);    // [4,32,4096]  2 MB
    float* psum    = (float*)(ws + 17825792);    // [2048,32]
    float* psumsq  = (float*)(ws + 18087936);    // [2048,32]
    float* bnAB    = (float*)(ws + 18350080);    // [2][32]

    pex_kernel<<<4096, 256, 0, stream>>>(x, x_out);
    peb_kernel<<<8192, 256, 0, stream>>>(bin, bpe);
    qk_kernel<<<dim3(64, BATCH), 64, 0, stream>>>(x_out, Wq, Wk, qf, kf);
    v_kernel<<<dim3(512, BATCH), 256, 0, stream>>>(bpe, Wv, vT);

    (void)hipFuncSetAttribute(reinterpret_cast<const void*>(attn_kernel),
                              hipFuncAttributeMaxDynamicSharedMemorySize,
                              ATTN_SMEM);
    attn_kernel<<<dim3(256, BATCH), 256, ATTN_SMEM, stream>>>(qf, kf, vT,
                                                              attn_out, oa);

    conv_kernel<<<dim3(512, BATCH), 256, 0, stream>>>(oa, Wc, bc, tconv,
                                                      psum, psumsq);
    bnfin_kernel<<<1, 256, 0, stream>>>(psum, psumsq, gamma, beta, bnAB);
    final_kernel<<<8192, 256, 0, stream>>>(tconv, bnAB, bpe, out_final);
}
